// MatchupLayer_39298950758568
// MI455X (gfx1250) — matrix operand order repaired and checked by reading, not compiled, not run on hardware
//
#include <hip/hip_runtime.h>
#include <hip/hip_bf16.h>

typedef __attribute__((ext_vector_type(16))) _Float16 v16h;
typedef __attribute__((ext_vector_type(8)))  _Float16 v8h;
typedef __attribute__((ext_vector_type(8)))  float    v8f;
typedef int v4i __attribute__((vector_size(16)));   // matches builtin param type

// Problem constants
#define BB 4096
#define NN 6
#define FF 512
#define CC 256
#define MM (BB*NN)          // 24576 rows per GEMM
#define KH 512              // K per half (W1 / W2)
#define RR 36               // N*N
#define CNT (BB*RR)         // BN population per channel: 147456

// Workspace layout (byte offsets, 256B-aligned regions)
#define OFF_W16   ((size_t)0)                         // _Float16[2][256][512] = 512 KB
#define OFF_A1    ((size_t)524288)                    // float[24576][256] = 25.17 MB
#define OFF_A2    (OFF_A1 + (size_t)MM*CC*4)
#define OFF_STATS (OFF_A2 + (size_t)MM*CC*4)          // float[512]: sum, sumsq
#define OFF_SS    (OFF_STATS + 2048)                  // float[512]: scale, shift

// CDNA5 async global->LDS path (ASYNCcnt), with portable fallback
#if defined(__has_builtin)
#if __has_builtin(__builtin_amdgcn_global_load_async_to_lds_b128) && \
    __has_builtin(__builtin_amdgcn_s_wait_asynccnt)
#define USE_ASYNC_LDS 1
#endif
#endif
#ifndef USE_ASYNC_LDS
#define USE_ASYNC_LDS 0
#endif

#define AS1 __attribute__((address_space(1)))
#define AS3 __attribute__((address_space(3)))

// LDS B-chunk row stride in halfs: 32 data + 8 pad = 80B rows ->
// start bank 20*c mod 64: 16 distinct 4-dword-aligned slots per half-wave.
#define BSTR 40

// ---------------------------------------------------------------------------
// Kernel 0: convert conv_w (C x 2F, f32) into two f16 matrices W1/W2
//           [s][c][k] with k<512; also zero the stats accumulators.
__global__ void __launch_bounds__(256) prep_kernel(
    const float* __restrict__ conv_w, _Float16* __restrict__ w16,
    float* __restrict__ stats) {
  const int idx = blockIdx.x * 256 + threadIdx.x;   // 0 .. 262143
  const int c  = idx >> 10;                         // /1024
  const int k  = idx & 1023;
  const int s  = k >> 9;
  const int kk = k & 511;
  w16[(((size_t)s * CC + c) << 9) + kk] = (_Float16)conv_w[idx];
  if (idx < 512) stats[idx] = 0.0f;
}

// ---------------------------------------------------------------------------
// Cooperative load of one B chunk (256 cols x 32 k halfs = 16KB) into LDS.
// Thread t loads column c = t: 64B from global -> padded LDS row.
__device__ __forceinline__ void load_b_chunk(const _Float16* __restrict__ wbase,
                                             _Float16* sdst, int k0, int t) {
  const _Float16* src = wbase + ((size_t)t << 9) + k0;  // w16 row t, 32 halfs
  _Float16* dst = sdst + t * BSTR;
#if USE_ASYNC_LDS
#pragma unroll
  for (int i = 0; i < 4; ++i) {
    __builtin_amdgcn_global_load_async_to_lds_b128(
        (AS1 v4i*)(src + i * 8), (AS3 v4i*)(dst + i * 8), 0, 0);
  }
#else
#pragma unroll
  for (int i = 0; i < 4; ++i) {
    *(v8h*)(dst + i * 8) = *(const v8h*)(src + i * 8);
  }
#endif
}

__device__ __forceinline__ void wait_b_chunk() {
#if USE_ASYNC_LDS
  __builtin_amdgcn_s_wait_asynccnt(0);
#endif
  __syncthreads();
}

// ---------------------------------------------------------------------------
// Kernel 1: A{1,2}[m][c] = sum_k p{1,2}[m][k] * W{1,2}[c][k]
// 8 waves / block, one 16-row tile per wave across all 256 output columns.
// B staged through double-buffered LDS via async global->LDS DMA (shared by
// all 8 waves); B fragments and next-K A fragments software-pipelined so
// DS / global latency overlaps WMMA execution.
__global__ void __launch_bounds__(256) gemm_kernel(
    const float* __restrict__ x, const _Float16* __restrict__ w16,
    float* __restrict__ a1, float* __restrict__ a2) {
  __shared__ _Float16 sB[2][CC * BSTR];   // 2 x 20KB

  const int t    = threadIdx.x;
  const int lane = t & 31;
  const int wv   = t >> 5;                 // 0..7
  const int s    = blockIdx.y;             // 0: p1/W1, 1: p2/W2
  const int tm   = blockIdx.x * 8 + wv;    // 0..1535 row tiles
  const int r    = lane & 15;

  // A-fragment source row: m = tm*16 + r maps to x[b, s, n, :]
  const int m = tm * 16 + r;
  const int b = m / 6;
  const int n = m - b * 6;
  const float* __restrict__ xrow = x + ((size_t)(b * 2 + s) * NN + n) * FF;
  const _Float16* __restrict__ wbase = w16 + ((size_t)s * CC) * KH;

  // CDNA5 16-bit A layout: lanes<16 hold K {0..7,16..23}, lanes>=16 {8..15,24..31}
  const int kk0 = (lane < 16) ? 0 : 8;
  // CDNA5 16-bit B layout: lanes<16 hold K {0..7,16..23}, lanes>=16 {8..15,24..31}
  const int kb0 = (lane < 16) ? 0 : 8;

  v8f acc[16];
  const v8f z = {0.f,0.f,0.f,0.f,0.f,0.f,0.f,0.f};
#pragma unroll
  for (int tt = 0; tt < 16; ++tt) acc[tt] = z;

  load_b_chunk(wbase, &sB[0][0], 0, t);

  // Prologue: A fragment for ks=0 in flight while B chunk 0 lands.
  float4 f0 = *(const float4*)(xrow + kk0);
  float4 f1 = *(const float4*)(xrow + kk0 + 4);
  float4 f2 = *(const float4*)(xrow + kk0 + 16);
  float4 f3 = *(const float4*)(xrow + kk0 + 20);

  wait_b_chunk();

  for (int ks = 0; ks < 16; ++ks) {
    const int k0  = ks * 32;
    const int cur = ks & 1;
    if (ks + 1 < 16) load_b_chunk(wbase, &sB[1 - cur][0], k0 + 32, t);

    // Prefetch next K-step's A fragment (consumed after the barrier below).
    float4 g0 = f0, g1 = f1, g2 = f2, g3 = f3;
    if (ks + 1 < 16) {
      const float* xn = xrow + k0 + 32 + kk0;
      g0 = *(const float4*)(xn);
      g1 = *(const float4*)(xn + 4);
      g2 = *(const float4*)(xn + 16);
      g3 = *(const float4*)(xn + 20);
    }

    // Convert current A fragment (loaded one iteration ago) to f16.
    v16h a;
    a[0]=(_Float16)f0.x;  a[1]=(_Float16)f0.y;  a[2]=(_Float16)f0.z;  a[3]=(_Float16)f0.w;
    a[4]=(_Float16)f1.x;  a[5]=(_Float16)f1.y;  a[6]=(_Float16)f1.z;  a[7]=(_Float16)f1.w;
    a[8]=(_Float16)f2.x;  a[9]=(_Float16)f2.y;  a[10]=(_Float16)f2.z; a[11]=(_Float16)f2.w;
    a[12]=(_Float16)f3.x; a[13]=(_Float16)f3.y; a[14]=(_Float16)f3.z; a[15]=(_Float16)f3.w;

    // Tile loop with explicit ping-pong B-fragment registers: the DS loads
    // for tile tt+1 target the *other* buffer than the one WMMA tt reads,
    // so loads stay outstanding across the WMMA (partial dscnt waits).
    const _Float16* sbuf = &sB[cur][0];
    v8h lo[2], hi[2];
    lo[0] = *(const v8h*)(sbuf + r * BSTR + kb0);
    hi[0] = *(const v8h*)(sbuf + r * BSTR + kb0 + 16);
#pragma unroll
    for (int tt = 0; tt < 16; ++tt) {
      const int p = tt & 1;
      if (tt + 1 < 16) {
        const int c = (tt + 1) * 16 + r;
        lo[1 - p] = *(const v8h*)(sbuf + c * BSTR + kb0);
        hi[1 - p] = *(const v8h*)(sbuf + c * BSTR + kb0 + 16);
      }
      const v16h bf = __builtin_shufflevector(lo[p], hi[p],
          0,1,2,3,4,5,6,7,8,9,10,11,12,13,14,15);
      acc[tt] = __builtin_amdgcn_wmma_f32_16x16x32_f16(
          false, a, false, bf, (short)0, acc[tt], false, false);
    }

    f0 = g0; f1 = g1; f2 = g2; f3 = g3;
    wait_b_chunk();   // next buffer ready; everyone done reading current
  }

  // C/D layout: VGPR v -> row v + 8*(lane>=16); col = lane&15
  float* __restrict__ aout = (s == 0) ? a1 : a2;
  const int rowhi = (lane >> 4) * 8;
#pragma unroll
  for (int tt = 0; tt < 16; ++tt) {
    const int col = tt * 16 + r;
#pragma unroll
    for (int v = 0; v < 8; ++v) {
      aout[(size_t)(tm * 16 + rowhi + v) * CC + col] = acc[tt][v];
    }
  }
}

// ---------------------------------------------------------------------------
// Kernel 2: per-channel sum / sumsq of relu(A1[b,j,c] + A2[b,j2,c] + bias[c])
// over all 36 (j,j2) pairs (equivalent to all (i,j): j2 = (j-i) mod 6).
__global__ void __launch_bounds__(256) stats_kernel(
    const float* __restrict__ a1, const float* __restrict__ a2,
    const float* __restrict__ conv_b, float* __restrict__ stats) {
  const int c = threadIdx.x;
  const float bias = conv_b[c];
  float sum = 0.f, ssq = 0.f;
  const int b0 = blockIdx.x * 8;
  for (int bb = 0; bb < 8; ++bb) {
    const int b = b0 + bb;
    float v1[6], v2[6];
#pragma unroll
    for (int j = 0; j < 6; ++j) {
      v1[j] = a1[(((size_t)b * 6 + j) << 8) + c] + bias;
      v2[j] = a2[(((size_t)b * 6 + j) << 8) + c];
    }
#pragma unroll
    for (int j = 0; j < 6; ++j)
#pragma unroll
      for (int j2 = 0; j2 < 6; ++j2) {
        const float h = fmaxf(v1[j] + v2[j2], 0.f);
        sum += h;
        ssq += h * h;
      }
  }
  atomicAdd(&stats[c], sum);
  atomicAdd(&stats[CC + c], ssq);
}

// ---------------------------------------------------------------------------
// Kernel 3: fold BN stats + gamma/beta into per-channel scale/shift.
__global__ void finalize_kernel(
    const float* __restrict__ stats, const float* __restrict__ gamma,
    const float* __restrict__ beta, float* __restrict__ ss) {
  const int c = threadIdx.x;
  const float inv = 1.0f / (float)CNT;
  const float mean = stats[c] * inv;
  const float var  = stats[CC + c] * inv - mean * mean;
  const float rstd = rsqrtf(var + 1e-5f);
  const float sc = gamma[c] * rstd;
  ss[c] = sc;
  ss[CC + c] = beta[c] - mean * sc;
}

// ---------------------------------------------------------------------------
// Kernel 4: out[b,c,i*6+j] = relu(A1[b,j,c]+A2[b,(j-i)%6,c]+bias)*scale+shift
// A1/A2 for batch b staged in LDS (padded stride 264 -> conflict-free),
// output written fully coalesced (256 consecutive floats per store wave).
#define SSTR 264
__global__ void __launch_bounds__(256) out_kernel(
    const float* __restrict__ a1, const float* __restrict__ a2,
    const float* __restrict__ conv_b, const float* __restrict__ ss,
    float* __restrict__ out) {
  __shared__ float sA1[6 * SSTR];
  __shared__ float sA2[6 * SSTR];
  __shared__ float sBias[CC], sScale[CC], sShift[CC];
  const int t = threadIdx.x;
  const int b = blockIdx.x;
#pragma unroll
  for (int j = 0; j < 6; ++j) {
    sA1[j * SSTR + t] = a1[(((size_t)b * 6 + j) << 8) + t];
    sA2[j * SSTR + t] = a2[(((size_t)b * 6 + j) << 8) + t];
  }
  sBias[t]  = conv_b[t];
  sScale[t] = ss[t];
  sShift[t] = ss[CC + t];
  __syncthreads();

  float* __restrict__ outb = out + (size_t)b * (CC * RR);
#pragma unroll 4
  for (int it = 0; it < 36; ++it) {
    const int o  = it * 256 + t;       // 0 .. 9215
    const int c  = o / 36;
    const int rr = o - c * 36;
    const int i  = rr / 6;
    const int j  = rr - i * 6;
    int j2 = j - i; if (j2 < 0) j2 += 6;
    float h = sA1[j * SSTR + c] + sA2[j2 * SSTR + c] + sBias[c];
    h = fmaxf(h, 0.f);
    outb[o] = h * sScale[c] + sShift[c];
  }
}

// ---------------------------------------------------------------------------
extern "C" void kernel_launch(void* const* d_in, const int* in_sizes, int n_in,
                              void* d_out, int out_size, void* d_ws, size_t ws_size,
                              hipStream_t stream) {
  const float* x      = (const float*)d_in[0];
  const float* conv_w = (const float*)d_in[1];
  const float* conv_b = (const float*)d_in[2];
  const float* gamma  = (const float*)d_in[3];
  const float* beta   = (const float*)d_in[4];
  float* out = (float*)d_out;

  char* ws = (char*)d_ws;
  _Float16* w16 = (_Float16*)(ws + OFF_W16);
  float* a1    = (float*)(ws + OFF_A1);
  float* a2    = (float*)(ws + OFF_A2);
  float* stats = (float*)(ws + OFF_STATS);
  float* ss    = (float*)(ws + OFF_SS);

  prep_kernel<<<1024, 256, 0, stream>>>(conv_w, w16, stats);
  gemm_kernel<<<dim3(MM / 16 / 8, 2), 256, 0, stream>>>(x, w16, a1, a2);
  stats_kernel<<<BB / 8, 256, 0, stream>>>(a1, a2, conv_b, stats);
  finalize_kernel<<<1, 256, 0, stream>>>(stats, gamma, beta, ss);
  out_kernel<<<BB, 256, 0, stream>>>(a1, a2, conv_b, ss, out);
}
